// TemporalSparseTrifocusedAttention_14413910245672
// MI455X (gfx1250) — matrix operand order repaired and checked by reading, not compiled, not run on hardware
//
#include <hip/hip_runtime.h>

typedef __bf16 bf16;
typedef __attribute__((ext_vector_type(8)))  bf16  v8bf;
typedef __attribute__((ext_vector_type(16))) bf16  v16bf;
typedef __attribute__((ext_vector_type(8)))  float v8f;

#define S_LEN  2048
#define BATCH  2
#define DMODEL 768
#define HEADS  12
#define DKDIM  64
#define BSROWS 4096       // BATCH * S_LEN
#define TOPK   409        // max(1, int(0.2*2048))
#define NEGV   (-1.0e9f)

#define WMMA_BF16(A_, B_, C_) \
  __builtin_amdgcn_wmma_f32_16x16x32_bf16(false, (A_), false, (B_), (short)0, (C_), false, false)

static __device__ __forceinline__ v16bf make16(v8bf lo, v8bf hi) {
  v16bf r;
#pragma unroll
  for (int i = 0; i < 8; ++i) { r[i] = lo[i]; r[i + 8] = hi[i]; }
  return r;
}
// reduce across the 16 lanes of a half-wave (xor masks 1..8 stay inside the half)
static __device__ __forceinline__ float grp_max(float v) {
#pragma unroll
  for (int m = 1; m < 16; m <<= 1) v = fmaxf(v, __shfl_xor(v, m, 32));
  return v;
}
static __device__ __forceinline__ float grp_sum(float v) {
#pragma unroll
  for (int m = 1; m < 16; m <<= 1) v += __shfl_xor(v, m, 32);
  return v;
}

// ---------------------------------------------------------------------------
// x -> bf16, and x + positional encoding -> bf16
// ---------------------------------------------------------------------------
__global__ void convert_x_kernel(const float* __restrict__ x,
                                 bf16* __restrict__ Xbf, bf16* __restrict__ XPbf) {
  const size_t total = (size_t)BSROWS * DMODEL;
  for (size_t i = (size_t)blockIdx.x * blockDim.x + threadIdx.x; i < total;
       i += (size_t)gridDim.x * blockDim.x) {
    int col = (int)(i % DMODEL);
    int srow = (int)((i / DMODEL) & (S_LEN - 1));
    float xv = x[i];
    Xbf[i] = (bf16)xv;
    int j2 = col & ~1;
    float div = __expf(-9.210340371976184f * (float)j2 / (float)DMODEL);
    float ang = (float)srow * div;
    float pe = (col & 1) ? __cosf(ang) : __sinf(ang);
    XPbf[i] = (bf16)(xv + pe);
  }
}

// ---------------------------------------------------------------------------
// W[K][N] fp32 -> Wt[N][K] bf16 (transposed so GEMM B-frags are contiguous)
// ---------------------------------------------------------------------------
__global__ void convert_wT_kernel(const float* __restrict__ W, bf16* __restrict__ Wt,
                                  int K, int N) {
  const size_t total = (size_t)K * N;
  for (size_t i = (size_t)blockIdx.x * blockDim.x + threadIdx.x; i < total;
       i += (size_t)gridDim.x * blockDim.x) {
    int k = (int)(i / N);
    int n = (int)(i % N);
    Wt[(size_t)n * K + k] = (bf16)W[i];
  }
}

// ---------------------------------------------------------------------------
// gate: fw = softmax(mean_s(x) @ w_gate + b_gate)   (one block per batch)
// ---------------------------------------------------------------------------
__global__ void gate_kernel(const float* __restrict__ x, const float* __restrict__ w_gate,
                            const float* __restrict__ b_gate, float* __restrict__ fw) {
  int b = blockIdx.x, tid = threadIdx.x;
  __shared__ float mean[DMODEL];
  for (int c = tid; c < DMODEL; c += blockDim.x) {
    float acc = 0.f;
    for (int i = 0; i < S_LEN; ++i)
      acc += x[((size_t)(b * S_LEN + i)) * DMODEL + c];
    mean[c] = acc * (1.0f / (float)S_LEN);
  }
  __syncthreads();
  if (tid == 0) {
    float z[3];
    for (int g = 0; g < 3; ++g) {
      float a = b_gate[g];
      for (int c = 0; c < DMODEL; ++c) a += mean[c] * w_gate[c * 3 + g];
      z[g] = a;
    }
    float m = fmaxf(z[0], fmaxf(z[1], z[2]));
    float e0 = __expf(z[0] - m), e1 = __expf(z[1] - m), e2 = __expf(z[2] - m);
    float inv = 1.0f / (e0 + e1 + e2);
    fw[b * 3 + 0] = e0 * inv; fw[b * 3 + 1] = e1 * inv; fw[b * 3 + 2] = e2 * inv;
  }
}

// ---------------------------------------------------------------------------
// sparse mask: importance = x @ w_sparse + b; top-409 per batch via bitonic sort
// ---------------------------------------------------------------------------
__global__ void sparse_topk_kernel(const float* __restrict__ x,
                                   const float* __restrict__ w_sparse,
                                   const float* __restrict__ b_sparse,
                                   unsigned char* __restrict__ mask) {
  int b = blockIdx.x, tid = threadIdx.x;
  __shared__ float vals[S_LEN];
  __shared__ int   idxs[S_LEN];
  for (int i = tid; i < S_LEN; i += blockDim.x) {
    float acc = 0.f;
    const float* xr = x + ((size_t)(b * S_LEN + i)) * DMODEL;
    for (int c = 0; c < DMODEL; ++c) acc += xr[c] * w_sparse[c];
    vals[i] = acc + b_sparse[0];
    idxs[i] = i;
    mask[b * S_LEN + i] = 0;
  }
  __syncthreads();
  for (int ksz = 2; ksz <= S_LEN; ksz <<= 1) {
    for (int j = ksz >> 1; j > 0; j >>= 1) {
      for (int t = tid; t < S_LEN; t += blockDim.x) {
        int ixj = t ^ j;
        if (ixj > t) {
          bool desc = ((t & ksz) == 0);
          float a = vals[t], c = vals[ixj];
          bool sw = desc ? (a < c) : (a > c);
          if (sw) {
            vals[t] = c; vals[ixj] = a;
            int ti = idxs[t]; idxs[t] = idxs[ixj]; idxs[ixj] = ti;
          }
        }
      }
      __syncthreads();
    }
  }
  for (int i = tid; i < TOPK; i += blockDim.x) mask[b * S_LEN + idxs[i]] = 1;
}

// ---------------------------------------------------------------------------
// generic bf16 WMMA GEMM: C[M,N] = A[M,K] (row-major bf16) @ Bt[N,K]^T (bf16)
// out_mode: 0 = fp32 (+bias), 1 = bf16 row-major, 2 = bf16 transposed (Ct[N][ldc])
// block = 128 (4 waves); each wave computes a 32x64 tile: two A fragments
// share every B fragment -> 8 WMMAs per 12 b128 loads per k-step.
// ---------------------------------------------------------------------------
__global__ void gemm_bf16_kernel(const bf16* __restrict__ A, int lda,
                                 const bf16* __restrict__ Bt, int ldb,
                                 void* __restrict__ Cout, int ldc,
                                 int M, int N, int K, int out_mode,
                                 const float* __restrict__ bias) {
  int wave = threadIdx.x >> 5;
  int lane = threadIdx.x & 31;
  int lm = lane & 15, hl = lane >> 4;
  int mbase = (blockIdx.x * 4 + wave) * 32;
  int nbase = blockIdx.y * 64;
  if (mbase >= M || nbase >= N) return;

  v8f acc[2][4] = {};
  const bf16* arow0 = A + (size_t)(mbase + lm) * lda;
  const bf16* arow1 = A + (size_t)(mbase + 16 + lm) * lda;
  for (int kb = 0; kb < K; kb += 32) {
    if (kb + 64 < K) {
      __builtin_prefetch(arow0 + kb + 64, 0, 3);
      __builtin_prefetch(arow1 + kb + 64, 0, 3);
    }
    v16bf af0 = make16(*(const v8bf*)(arow0 + kb + hl * 8),
                       *(const v8bf*)(arow0 + kb + 16 + hl * 8));
    v16bf af1 = make16(*(const v8bf*)(arow1 + kb + hl * 8),
                       *(const v8bf*)(arow1 + kb + 16 + hl * 8));
#pragma unroll
    for (int j = 0; j < 4; ++j) {
      const bf16* bp = Bt + (size_t)(nbase + j * 16 + lm) * ldb + kb + hl * 8;
      v16bf bfv = make16(*(const v8bf*)bp, *(const v8bf*)(bp + 16));
      acc[0][j] = WMMA_BF16(af0, bfv, acc[0][j]);
      acc[1][j] = WMMA_BF16(af1, bfv, acc[1][j]);
    }
  }

  if (out_mode == 0) {
    float* C = (float*)Cout;
#pragma unroll
    for (int mi = 0; mi < 2; ++mi) {
#pragma unroll
      for (int j = 0; j < 4; ++j) {
        int col = nbase + j * 16 + lm;
        float bv = bias ? bias[col] : 0.0f;
#pragma unroll
        for (int r = 0; r < 8; ++r)
          C[(size_t)(mbase + mi * 16 + r + hl * 8) * ldc + col] = acc[mi][j][r] + bv;
      }
    }
  } else if (out_mode == 1) {
    bf16* C = (bf16*)Cout;
#pragma unroll
    for (int mi = 0; mi < 2; ++mi) {
#pragma unroll
      for (int j = 0; j < 4; ++j) {
        int col = nbase + j * 16 + lm;
#pragma unroll
        for (int r = 0; r < 8; ++r)
          C[(size_t)(mbase + mi * 16 + r + hl * 8) * ldc + col] = (bf16)acc[mi][j][r];
      }
    }
  } else { // transposed bf16: Ct[n][m], contiguous 8-element store per reg group
    bf16* Ct = (bf16*)Cout;
#pragma unroll
    for (int mi = 0; mi < 2; ++mi) {
#pragma unroll
      for (int j = 0; j < 4; ++j) {
        v8bf t;
#pragma unroll
        for (int r = 0; r < 8; ++r) t[r] = (bf16)acc[mi][j][r];
        *(v8bf*)(Ct + (size_t)(nbase + j * 16 + lm) * ldc + mbase + mi * 16 + hl * 8) = t;
      }
    }
  }
}

// ---------------------------------------------------------------------------
// flash attention, one wave per (q-tile 16, head, batch)
// mode: 0 = sparse column mask, 1 = local band (|rel|<=8), 2 = temporal (col%256==0)
// Q,K: bf16 [b][s][768] (head-major cols); Vt: bf16 [head*64][b*s]
// ---------------------------------------------------------------------------
__global__ void attn_kernel(const bf16* __restrict__ Q, const bf16* __restrict__ Kb,
                            const bf16* __restrict__ Vt,
                            const unsigned char* __restrict__ smask,
                            float* __restrict__ O, int mode) {
  int qt = blockIdx.x, h = blockIdx.y, b = blockIdx.z;
  int lane = threadIdx.x;
  int lm = lane & 15, hl = lane >> 4;
  int qbase = qt * 16;

  __shared__ __align__(16) bf16 Plds[16 * 32];

  const bf16* qp = Q + ((size_t)(b * S_LEN + qbase + lm)) * DMODEL + h * DKDIM;
  v16bf qa0 = make16(*(const v8bf*)(qp + hl * 8), *(const v8bf*)(qp + 16 + hl * 8));
  v16bf qa1 = make16(*(const v8bf*)(qp + 32 + hl * 8), *(const v8bf*)(qp + 48 + hl * 8));

  float mi[8], li[8];
#pragma unroll
  for (int r = 0; r < 8; ++r) { mi[r] = -1.0e30f; li[r] = 0.0f; }
  v8f acc[4] = {};

  int g0 = 0, g1 = S_LEN / 32;
  if (mode == 1) { // local band: keys in [qbase-8, qbase+23]
    int klo = qbase - 8; if (klo < 0) klo = 0;
    int khi = qbase + 23; if (khi > S_LEN - 1) khi = S_LEN - 1;
    g0 = klo >> 5; g1 = (khi >> 5) + 1;
  }

  for (int g = g0; g < g1; ++g) {
    if (mode == 2 && (g & 7)) continue; // temporal: only groups containing col%256==0

    int k0 = g << 5;
    v8f sc[2];
#pragma unroll
    for (int t = 0; t < 2; ++t) {
      const bf16* kp = Kb + ((size_t)(b * S_LEN + k0 + t * 16 + lm)) * DMODEL + h * DKDIM;
      v16bf b0 = make16(*(const v8bf*)(kp + hl * 8), *(const v8bf*)(kp + 16 + hl * 8));
      v16bf b1 = make16(*(const v8bf*)(kp + 32 + hl * 8), *(const v8bf*)(kp + 48 + hl * 8));
      v8f c = {};
      c = WMMA_BF16(qa0, b0, c);
      c = WMMA_BF16(qa1, b1, c);
      sc[t] = c;
    }

    // scale + mask
#pragma unroll
    for (int t = 0; t < 2; ++t) {
      int col = k0 + t * 16 + lm;
      bool okc = true;
      if (mode == 0) okc = smask[b * S_LEN + col] != 0;
      else if (mode == 2) okc = (col & 255) == 0;
#pragma unroll
      for (int r = 0; r < 8; ++r) {
        float v = sc[t][r] * 0.125f; // 1/sqrt(64)
        bool ok = okc;
        if (mode == 1) { int rel = col - (qbase + r + hl * 8); ok = (rel >= -8) && (rel <= 8); }
        sc[t][r] = ok ? v : NEGV;
      }
    }

    // online softmax (row-wise across the 16 lanes of each half-wave)
    float p0[8], p1[8];
#pragma unroll
    for (int r = 0; r < 8; ++r) {
      float mloc = grp_max(fmaxf(sc[0][r], sc[1][r]));
      float mnew = fmaxf(mi[r], mloc);
      float alpha = __expf(mi[r] - mnew);
      p0[r] = __expf(sc[0][r] - mnew);
      p1[r] = __expf(sc[1][r] - mnew);
      float rs = grp_sum(p0[r] + p1[r]);
      li[r] = li[r] * alpha + rs;
      mi[r] = mnew;
#pragma unroll
      for (int j = 0; j < 4; ++j) acc[j][r] *= alpha;
    }

    // D-layout P -> LDS (row-major 16x32) -> A-layout frag
#pragma unroll
    for (int r = 0; r < 8; ++r) {
      int prow = r + hl * 8;
      Plds[prow * 32 + lm]      = (bf16)p0[r];
      Plds[prow * 32 + 16 + lm] = (bf16)p1[r];
    }
    asm volatile("s_wait_dscnt 0x0" ::: "memory");
    v16bf pa = make16(*(const v8bf*)&Plds[lm * 32 + hl * 8],
                      *(const v8bf*)&Plds[lm * 32 + 16 + hl * 8]);

    // P(16x32) @ V(32x64): 4 WMMAs, keys contiguous along Vt rows
#pragma unroll
    for (int j = 0; j < 4; ++j) {
      const bf16* vp = Vt + ((size_t)(h * DKDIM + j * 16 + lm)) * BSROWS + b * S_LEN + k0 + hl * 8;
      v16bf vb = make16(*(const v8bf*)vp, *(const v8bf*)(vp + 16));
      acc[j] = WMMA_BF16(pa, vb, acc[j]);
    }
  }

#pragma unroll
  for (int j = 0; j < 4; ++j) {
    int col = h * DKDIM + j * 16 + lm;
#pragma unroll
    for (int r = 0; r < 8; ++r) {
      int row = qbase + r + hl * 8;
      O[((size_t)(b * S_LEN + row)) * DMODEL + col] = acc[j][r] / li[r];
    }
  }
}

// ---------------------------------------------------------------------------
// LayerNorm + gate scale, writes bf16 slice of the concat buffer [BS][2304]
// one block (256 thr) per row
// ---------------------------------------------------------------------------
__global__ void ln_gate_kernel(const float* __restrict__ O, const float* __restrict__ gamma,
                               const float* __restrict__ beta, const float* __restrict__ fw,
                               int branch, bf16* __restrict__ comb) {
  int row = blockIdx.x, tid = threadIdx.x;
  const float* orow = O + (size_t)row * DMODEL;
  __shared__ float red[256];

  float s = 0.f;
  for (int c = tid; c < DMODEL; c += 256) s += orow[c];
  red[tid] = s; __syncthreads();
  for (int st = 128; st > 0; st >>= 1) { if (tid < st) red[tid] += red[tid + st]; __syncthreads(); }
  float mu = red[0] * (1.0f / (float)DMODEL);
  __syncthreads();

  float v = 0.f;
  for (int c = tid; c < DMODEL; c += 256) { float d = orow[c] - mu; v += d * d; }
  red[tid] = v; __syncthreads();
  for (int st = 128; st > 0; st >>= 1) { if (tid < st) red[tid] += red[tid + st]; __syncthreads(); }
  float var = red[0] * (1.0f / (float)DMODEL);
  float rs = rsqrtf(var + 1e-5f);
  float gwt = fw[(row >> 11) * 3 + branch];

  for (int c = tid; c < DMODEL; c += 256) {
    float val = (orow[c] - mu) * rs * gamma[c] + beta[c];
    comb[(size_t)row * (3 * DMODEL) + branch * DMODEL + c] = (bf16)(val * gwt);
  }
}

// ---------------------------------------------------------------------------
extern "C" void kernel_launch(void* const* d_in, const int* in_sizes, int n_in,
                              void* d_out, int out_size, void* d_ws, size_t ws_size,
                              hipStream_t stream) {
  (void)in_sizes; (void)n_in; (void)out_size; (void)ws_size;
  const float* x        = (const float*)d_in[0];
  const float* wproj[9] = { (const float*)d_in[1], (const float*)d_in[2], (const float*)d_in[3],
                            (const float*)d_in[4], (const float*)d_in[5], (const float*)d_in[6],
                            (const float*)d_in[7], (const float*)d_in[8], (const float*)d_in[9] };
  const float* w_out    = (const float*)d_in[10];
  const float* b_out    = (const float*)d_in[11];
  const float* w_gate   = (const float*)d_in[12];
  const float* b_gate   = (const float*)d_in[13];
  const float* w_sparse = (const float*)d_in[14];
  const float* b_sparse = (const float*)d_in[15];
  const float* gammas[3] = { (const float*)d_in[16], (const float*)d_in[18], (const float*)d_in[20] };
  const float* betas[3]  = { (const float*)d_in[17], (const float*)d_in[19], (const float*)d_in[21] };

  char* ws = (char*)d_ws;
  size_t off = 0;
  auto take = [&](size_t bytes) -> char* {
    char* p = ws + off;
    off += (bytes + 255) & ~(size_t)255;
    return p;
  };
  bf16*  Xbf   = (bf16*)take((size_t)BSROWS * DMODEL * 2);
  bf16*  XPbf  = (bf16*)take((size_t)BSROWS * DMODEL * 2);
  bf16*  Wq    = (bf16*)take((size_t)DMODEL * DMODEL * 2);
  bf16*  Wk    = (bf16*)take((size_t)DMODEL * DMODEL * 2);
  bf16*  Wv    = (bf16*)take((size_t)DMODEL * DMODEL * 2);
  bf16*  Qb    = (bf16*)take((size_t)BSROWS * DMODEL * 2);
  bf16*  Kbuf  = (bf16*)take((size_t)BSROWS * DMODEL * 2);
  bf16*  Vt    = (bf16*)take((size_t)DMODEL * BSROWS * 2);
  float* Obuf  = (float*)take((size_t)BSROWS * DMODEL * 4);
  bf16*  Comb  = (bf16*)take((size_t)BSROWS * 3 * DMODEL * 2);
  bf16*  WtOut = (bf16*)take((size_t)DMODEL * 3 * DMODEL * 2);
  unsigned char* maskp = (unsigned char*)take((size_t)BATCH * S_LEN);
  float* fwp   = (float*)take(32);

  convert_x_kernel<<<2048, 256, 0, stream>>>(x, Xbf, XPbf);
  gate_kernel<<<BATCH, 256, 0, stream>>>(x, w_gate, b_gate, fwp);
  sparse_topk_kernel<<<BATCH, 256, 0, stream>>>(x, w_sparse, b_sparse, maskp);
  convert_wT_kernel<<<2048, 256, 0, stream>>>(w_out, WtOut, 3 * DMODEL, DMODEL);

  dim3 gemm_grid(BSROWS / 128, DMODEL / 64);   // 32 x 12 blocks, 4 waves/block, 32x64 per wave
  for (int br = 0; br < 3; ++br) {
    const bf16* Xin = (br == 2) ? XPbf : Xbf;
    convert_wT_kernel<<<1024, 256, 0, stream>>>(wproj[br * 3 + 0], Wq, DMODEL, DMODEL);
    convert_wT_kernel<<<1024, 256, 0, stream>>>(wproj[br * 3 + 1], Wk, DMODEL, DMODEL);
    convert_wT_kernel<<<1024, 256, 0, stream>>>(wproj[br * 3 + 2], Wv, DMODEL, DMODEL);

    gemm_bf16_kernel<<<gemm_grid, 128, 0, stream>>>(Xin, DMODEL, Wq, DMODEL, Qb,   DMODEL,
                                                    BSROWS, DMODEL, DMODEL, 1, nullptr);
    gemm_bf16_kernel<<<gemm_grid, 128, 0, stream>>>(Xin, DMODEL, Wk, DMODEL, Kbuf, DMODEL,
                                                    BSROWS, DMODEL, DMODEL, 1, nullptr);
    gemm_bf16_kernel<<<gemm_grid, 128, 0, stream>>>(Xin, DMODEL, Wv, DMODEL, Vt,   BSROWS,
                                                    BSROWS, DMODEL, DMODEL, 2, nullptr);

    dim3 attn_grid(S_LEN / 16, HEADS, BATCH);
    attn_kernel<<<attn_grid, 32, 0, stream>>>(Qb, Kbuf, Vt, maskp, Obuf, br);

    ln_gate_kernel<<<BSROWS, 256, 0, stream>>>(Obuf, gammas[br], betas[br], fwp, br, Comb);
  }

  gemm_bf16_kernel<<<gemm_grid, 128, 0, stream>>>(Comb, 3 * DMODEL, WtOut, 3 * DMODEL,
                                                  d_out, DMODEL,
                                                  BSROWS, DMODEL, 3 * DMODEL, 0, b_out);
}
